// Block_13709535609166
// MI455X (gfx1250) — matrix operand order repaired and checked by reading, not compiled, not run on hardware
//
#include <hip/hip_runtime.h>
#include <hip/hip_bf16.h>

typedef __attribute__((ext_vector_type(16))) __bf16 v16bf;
typedef __attribute__((ext_vector_type(8)))  float  v8f;
typedef unsigned int u32x4 __attribute__((ext_vector_type(4)));
typedef int          i32x4 __attribute__((ext_vector_type(4)));
typedef int          i32x8 __attribute__((ext_vector_type(8)));

#define WMMA_BF16(A_, B_, C_) \
  __builtin_amdgcn_wmma_f32_16x16x32_bf16(false, (A_), false, (B_), (short)0, (C_), false, false)

#if defined(__HIP_DEVICE_COMPILE__) && __has_builtin(__builtin_amdgcn_tensor_load_to_lds)
#define HAVE_TDM 1
#else
#define HAVE_TDM 0
#endif

#define B_SZ     8
#define T_SEQ    1024
#define D_MODEL  1024
#define N_HEADS  16
#define HEAD_DIM 64
#define FF_DIM   4096
#define M_ROWS   (B_SZ * T_SEQ)   // 8192

// ---------------------------------------------------------------------------
// TDM: 2-D tile load Global -> LDS (D# per CDNA5 ISA ch.8). data_size=2B.
// LDS padding: +8 DWORDs after every 16 DWORDs => row stride 32+16 halves = 48.
// ---------------------------------------------------------------------------
#if HAVE_TDM
__device__ __forceinline__ void tdm_load_2d(unsigned lds_off, const void* gaddr,
                                            unsigned tensor_d0, unsigned tensor_d1,
                                            unsigned tile_d0, unsigned tile_d1,
                                            unsigned long long stride0) {
  unsigned long long ga = (unsigned long long)(uintptr_t)gaddr;
  u32x4 g0;
  g0[0] = 1u                       // count=1 descriptor
        | (1u << 20)               // pad_enable
        | (3u << 22)               // pad_interval: 16 DWORDs
        | (7u << 25);              // pad_amount:   8 DWORDs
  g0[1] = lds_off;                                            // lds_addr (bytes)
  g0[2] = (unsigned)(ga & 0xFFFFFFFFu);                       // global_addr lo
  g0[3] = (unsigned)((ga >> 32) & 0x01FFFFFFu) | (2u << 30);  // global_addr hi | type=2
  i32x8 g1;
  g1[0] = (int)(1u << 16);  // workgroup_mask=0, data_size=1 (2 bytes)
  g1[1] = (int)((tensor_d0 & 0xFFFFu) << 16);                               // dim0 lo
  g1[2] = (int)(((tensor_d0 >> 16) & 0xFFFFu) | ((tensor_d1 & 0xFFFFu) << 16));
  g1[3] = (int)(((tensor_d1 >> 16) & 0xFFFFu) | ((tile_d0 & 0xFFFFu) << 16));
  g1[4] = (int)(tile_d1 & 0xFFFFu);                                         // tile_dim2=0
  g1[5] = (int)(stride0 & 0xFFFFFFFFull);
  g1[6] = (int)((stride0 >> 32) & 0xFFFFull);                               // dim1_stride=0
  g1[7] = 0;
  i32x4 z4 = {0, 0, 0, 0};
#if __clang_major__ >= 23
  i32x8 z8 = {0, 0, 0, 0, 0, 0, 0, 0};
  __builtin_amdgcn_tensor_load_to_lds(g0, g1, z4, z4, z8, 0);
#else
  __builtin_amdgcn_tensor_load_to_lds(g0, g1, z4, z4, 0);
#endif
}
#endif

// ---------------------------------------------------------------------------
// Weight prep (one-time; amortized over 64 M-tiles of reuse):
//  - Wq/Wk/Wv (H,D,HD) f32 -> transposed bf16 [3*D][D]   (row n, cols k=d)
//  - generic [K][N] f32    -> transposed bf16 [N][K]
// ---------------------------------------------------------------------------
__global__ __launch_bounds__(256) void pack_qkv_wT_kernel(
    const float* __restrict__ Wq, const float* __restrict__ Wk,
    const float* __restrict__ Wv, __bf16* __restrict__ Wp) {
  int idx = blockIdx.x * 256 + threadIdx.x;  // over 3*1024*1024, d fastest
  int d = idx & 1023;
  int n = idx >> 10;                         // 0..3071
  int which = n >> 10;
  int h = (n >> 6) & 15;
  int e = n & 63;
  const float* W = (which == 0) ? Wq : (which == 1) ? Wk : Wv;
  Wp[idx] = (__bf16)W[h * 65536 + d * 64 + e];
}

__global__ __launch_bounds__(256) void pack_qkv_b_kernel(
    const float* __restrict__ bq, const float* __restrict__ bk,
    const float* __restrict__ bv, float* __restrict__ bp) {
  int n = blockIdx.x * 256 + threadIdx.x;
  if (n < 3072) {
    int which = n >> 10, j = n & 1023;
    const float* p = (which == 0) ? bq : (which == 1) ? bk : bv;
    bp[n] = p[j];
  }
}

__global__ __launch_bounds__(256) void transpose_cvt_kernel(
    const float* __restrict__ src, __bf16* __restrict__ dst, int K, int N) {
  __shared__ float tile[32][33];
  const int k0 = blockIdx.y * 32, n0 = blockIdx.x * 32;
  const int tx = threadIdx.x & 31, ty = threadIdx.x >> 5;  // 32 x 8
#pragma unroll
  for (int i = 0; i < 32; i += 8)
    tile[ty + i][tx] = src[(size_t)(k0 + ty + i) * N + n0 + tx];
  __syncthreads();
#pragma unroll
  for (int i = 0; i < 32; i += 8)
    dst[(size_t)(n0 + ty + i) * K + k0 + tx] = (__bf16)tile[tx][ty + i];
}

// ---------------------------------------------------------------------------
// LayerNorm: f32 [rows,1024] -> bf16
// ---------------------------------------------------------------------------
__global__ __launch_bounds__(256) void layernorm_kernel(
    const float* __restrict__ x, const float* __restrict__ g,
    const float* __restrict__ beta, __bf16* __restrict__ out) {
  const int row = blockIdx.x;
  const int tid = threadIdx.x;
  const float* xr = x + (size_t)row * D_MODEL;
  float4 v = ((const float4*)xr)[tid];
  float s  = v.x + v.y + v.z + v.w;
  float s2 = v.x * v.x + v.y * v.y + v.z * v.z + v.w * v.w;
#pragma unroll
  for (int o = 16; o > 0; o >>= 1) {
    s  += __shfl_xor(s,  o, 32);
    s2 += __shfl_xor(s2, o, 32);
  }
  __shared__ float red[2][8];
  if ((tid & 31) == 0) { red[0][tid >> 5] = s; red[1][tid >> 5] = s2; }
  __syncthreads();
  float S = 0.f, S2 = 0.f;
#pragma unroll
  for (int i = 0; i < 8; ++i) { S += red[0][i]; S2 += red[1][i]; }
  const float mu  = S * (1.0f / D_MODEL);
  const float var = S2 * (1.0f / D_MODEL) - mu * mu;
  const float r   = rsqrtf(var + 1e-5f);
  __bf16* orow = out + (size_t)row * D_MODEL;
  const int c = tid * 4;
  float4 gv = ((const float4*)g)[tid];
  float4 bv = ((const float4*)beta)[tid];
  orow[c + 0] = (__bf16)((v.x - mu) * r * gv.x + bv.x);
  orow[c + 1] = (__bf16)((v.y - mu) * r * gv.y + bv.y);
  orow[c + 2] = (__bf16)((v.z - mu) * r * gv.z + bv.z);
  orow[c + 3] = (__bf16)((v.w - mu) * r * gv.w + bv.w);
}

// ---------------------------------------------------------------------------
// bf16 WMMA GEMM: C[M,N] = A[M,K] * Bt[N,K]^T (+bias)(+epilogue)
// 128x128 / WG, BK=32, 8 waves (2x4), wave = 64x32 = 4x2 fragments.
// Tiles staged by the Tensor Data Mover (double-buffered, wave0 issues,
// s_wait_tensorcnt + barrier for handoff). Fallback: vector LDS copies.
// EP: 0 = QKV scatter (V transposed), 1 = +resid -> f32, 2 = GELU -> bf16
// ---------------------------------------------------------------------------
#define TILE_BYTES 12288          // 128 rows * 48 halves * 2
#define GEMM_SMEM  (4 * TILE_BYTES)

template <int EP>
__global__ __launch_bounds__(256) void gemm_bf16_kernel(
    const __bf16* __restrict__ A, const __bf16* __restrict__ Bt,
    const float* __restrict__ bias, int M, int N, int K,
    float* __restrict__ outF, __bf16* __restrict__ outB,
    const float* __restrict__ resid,
    __bf16* __restrict__ qOut, __bf16* __restrict__ kOut,
    __bf16* __restrict__ vtOut) {
  extern __shared__ char gsm[];  // [A0|B0|A1|B1], each 128x48 halves

  const int tid  = threadIdx.x;
  const int lane = tid & 31;
  const int wave = tid >> 5;
  const int wm   = wave >> 2;
  const int wn   = wave & 3;
  const int l15  = lane & 15;
  const int lh   = lane >> 4;
  const int bm   = blockIdx.y * 128;
  const int bn   = blockIdx.x * 128;

  v8f acc[4][2] = {};

#if HAVE_TDM
  if (wave == 0) {  // prologue: stage tile k0=0 into buffer 0
    tdm_load_2d(0,          A  + (size_t)bm * K, K, M, 32, 128, (unsigned long long)K);
    tdm_load_2d(TILE_BYTES, Bt + (size_t)bn * K, K, N, 32, 128, (unsigned long long)K);
  }
  int cur = 0;
  for (int k0 = 0; k0 < K; k0 += 32, cur ^= 1) {
    if (wave == 0) {
      if (k0 + 32 < K) {  // stage next tile into other buffer
        const unsigned bufOff = (unsigned)((cur ^ 1) * 2 * TILE_BYTES);
        tdm_load_2d(bufOff,              A  + (size_t)bm * K + k0 + 32, K, M, 32, 128,
                    (unsigned long long)K);
        tdm_load_2d(bufOff + TILE_BYTES, Bt + (size_t)bn * K + k0 + 32, K, N, 32, 128,
                    (unsigned long long)K);
        __builtin_amdgcn_s_wait_tensorcnt(2);  // current tile's 2 loads done
      } else {
        __builtin_amdgcn_s_wait_tensorcnt(0);
      }
    }
    __syncthreads();
    const __bf16* sA = (const __bf16*)(gsm + cur * 2 * TILE_BYTES);
    const __bf16* sB = (const __bf16*)(gsm + cur * 2 * TILE_BYTES + TILE_BYTES);
#else
  for (int k0 = 0; k0 < K; k0 += 32) {
    {  // fallback staging: 32B per thread per tile
      __bf16* dA = (__bf16*)gsm;
      __bf16* dB = (__bf16*)(gsm + TILE_BYTES);
      const int row = tid >> 1, colh = (tid & 1) * 16;
      const uint4* asrc = (const uint4*)(A + (size_t)(bm + row) * K + k0 + colh);
      *(uint4*)&dA[row * 48 + colh]     = asrc[0];
      *(uint4*)&dA[row * 48 + colh + 8] = asrc[1];
      const uint4* bsrc = (const uint4*)(Bt + (size_t)(bn + row) * K + k0 + colh);
      *(uint4*)&dB[row * 48 + colh]     = bsrc[0];
      *(uint4*)&dB[row * 48 + colh + 8] = bsrc[1];
    }
    __syncthreads();
    const __bf16* sA = (const __bf16*)gsm;
    const __bf16* sB = (const __bf16*)(gsm + TILE_BYTES);
#endif
    v16bf af[4], bfr[2];
    // A fragment, contiguous K: lanes 0-15 K{0..15}, 16-31 K{16..31}
#pragma unroll
    for (int r = 0; r < 4; ++r) {
      const __bf16* p = &sA[(wm * 64 + r * 16 + l15) * 48 + lh * 16];
      *(uint4*)&af[r]       = *(const uint4*)p;
      *((uint4*)&af[r] + 1) = *(const uint4*)(p + 8);
    }
    // B fragment: lane n = l15, K-halves contiguous in [N][K] tile
#pragma unroll
    for (int c = 0; c < 2; ++c) {
      const __bf16* p = &sB[(wn * 32 + c * 16 + l15) * 48 + lh * 16];
      *(uint4*)&bfr[c]       = *(const uint4*)p;
      *((uint4*)&bfr[c] + 1) = *(const uint4*)(p + 8);
    }
#pragma unroll
    for (int r = 0; r < 4; ++r)
#pragma unroll
      for (int c = 0; c < 2; ++c)
        acc[r][c] = WMMA_BF16(af[r], bfr[c], acc[r][c]);
    __syncthreads();
  }

  // Epilogue. C layout: lane col = l15, VGPR v -> row v + 8*lh.
#pragma unroll
  for (int r = 0; r < 4; ++r) {
#pragma unroll
    for (int c = 0; c < 2; ++c) {
#pragma unroll
      for (int vv = 0; vv < 8; ++vv) {
        const int m = bm + wm * 64 + r * 16 + vv + 8 * lh;
        const int n = bn + wn * 32 + c * 16 + l15;
        float val = acc[r][c][vv];
        if (bias) val += bias[n];
        if (EP == 0) {
          const int which = n >> 10;
          const int h = (n >> 6) & 15;
          const int e = n & 63;
          const int b = m >> 10;
          const int t = m & 1023;
          if (which == 2) {  // V stored transposed: [B,H,HD,T]
            vtOut[((size_t)(b * N_HEADS + h) * HEAD_DIM + e) * T_SEQ + t] = (__bf16)val;
          } else {
            __bf16* dst = (which == 0) ? qOut : kOut;
            dst[((size_t)(b * N_HEADS + h) * T_SEQ + t) * HEAD_DIM + e] = (__bf16)val;
          }
        } else if (EP == 1) {
          outF[(size_t)m * N + n] = val + resid[(size_t)m * N + n];
        } else {
          float gl = 0.5f * val * (1.0f + erff(val * 0.70710678118f));
          outB[(size_t)m * N + n] = (__bf16)gl;
        }
      }
    }
  }
}

// ---------------------------------------------------------------------------
// Attention: WG per (b*h, 32 Q rows). 32x1024 f32 scores + 32x1024 bf16 probs
// in 194 KB dynamic LDS (CDNA5 WGP: 320 KB). V read pre-transposed from
// global, so P*V B-fragments are contiguous 32B loads. Softmax normalization
// folded into the ctx store via per-row 1/sum.
// ---------------------------------------------------------------------------
#define SSTR 1032   // padded f32 row stride
#define PSTR 1040   // padded bf16 row stride
#define ATTN_SMEM (32 * SSTR * 4 + 32 * PSTR * 2 + 128)

__global__ __launch_bounds__(256) void attn_kernel(
    const __bf16* __restrict__ Q, const __bf16* __restrict__ K,
    const __bf16* __restrict__ Vt, __bf16* __restrict__ ctx) {
  extern __shared__ char smem[];
  float*  sS   = (float*)smem;                                  // [32][SSTR]
  __bf16* sP   = (__bf16*)(smem + 32 * SSTR * 4);               // [32][PSTR]
  float*  sInv = (float*)(smem + 32 * SSTR * 4 + 32 * PSTR * 2);// [32]

  const int bh  = blockIdx.y;
  const int t0  = blockIdx.x * 32;
  const int tid = threadIdx.x, lane = tid & 31, wave = tid >> 5;
  const int wm = wave >> 2, wn = wave & 3;
  const int l15 = lane & 15, lh = lane >> 4;

  const __bf16* Qb  = Q  + (size_t)bh * T_SEQ * HEAD_DIM;
  const __bf16* Kb  = K  + (size_t)bh * T_SEQ * HEAD_DIM;
  const __bf16* Vtb = Vt + (size_t)bh * HEAD_DIM * T_SEQ;

  // Q A-fragments direct from global
  v16bf qa[2];
  {
    const __bf16* qp = Qb + (size_t)(t0 + wm * 16 + l15) * HEAD_DIM + lh * 16;
#pragma unroll
    for (int kt = 0; kt < 2; ++kt) {
      *(uint4*)&qa[kt]       = *(const uint4*)(qp + kt * 32);
      *((uint4*)&qa[kt] + 1) = *(const uint4*)(qp + kt * 32 + 8);
    }
  }

  // Phase 1: S = Q K^T / 8 ; each wave covers 256 keys (16 fragments)
  for (int c = 0; c < 16; ++c) {
    const int s0 = wn * 256 + c * 16;
    const __bf16* kp = Kb + (size_t)(s0 + l15) * HEAD_DIM + lh * 16;
    if (c + 1 < 16)
      __builtin_prefetch(kp + 16 * HEAD_DIM, 0, 0);
    v16bf kb0, kb1;  // K^T B-fragments: contiguous 32B per lane
    *(uint4*)&kb0       = *(const uint4*)kp;
    *((uint4*)&kb0 + 1) = *(const uint4*)(kp + 8);
    *(uint4*)&kb1       = *(const uint4*)(kp + 32);
    *((uint4*)&kb1 + 1) = *(const uint4*)(kp + 40);
    v8f sacc = {};
    sacc = WMMA_BF16(qa[0], kb0, sacc);
    sacc = WMMA_BF16(qa[1], kb1, sacc);
#pragma unroll
    for (int vv = 0; vv < 8; ++vv)
      sS[(wm * 16 + vv + 8 * lh) * SSTR + s0 + l15] = sacc[vv] * 0.125f;
  }
  __syncthreads();

  // Phase 2: row max + exp (unnormalized bf16 P) + 1/sum table. 8 lanes/row.
  {
    const int row = tid >> 3, sub = tid & 7;
    const float4* rp4 = (const float4*)(sS + row * SSTR + sub * 128);
    float mx = -3.0e38f;
#pragma unroll 4
    for (int i = 0; i < 32; ++i) {
      float4 v = rp4[i];
      mx = fmaxf(mx, fmaxf(fmaxf(v.x, v.y), fmaxf(v.z, v.w)));
    }
#pragma unroll
    for (int o = 1; o < 8; o <<= 1) mx = fmaxf(mx, __shfl_xor(mx, o, 32));
    float sum = 0.f;
    __bf16* pp = sP + row * PSTR + sub * 128;
#pragma unroll 4
    for (int i = 0; i < 32; ++i) {
      float4 v = rp4[i];
      float e0 = __expf(v.x - mx), e1 = __expf(v.y - mx);
      float e2 = __expf(v.z - mx), e3 = __expf(v.w - mx);
      sum += (e0 + e1) + (e2 + e3);
      pp[i * 4 + 0] = (__bf16)e0;
      pp[i * 4 + 1] = (__bf16)e1;
      pp[i * 4 + 2] = (__bf16)e2;
      pp[i * 4 + 3] = (__bf16)e3;
    }
#pragma unroll
    for (int o = 1; o < 8; o <<= 1) sum += __shfl_xor(sum, o, 32);
    if (sub == 0) sInv[row] = 1.0f / sum;
  }
  __syncthreads();

  // Phase 3: ctx = (P V) * inv. Wave (wm,wn): rows wm*16..+16, e cols wn*16..+16
  v8f cacc = {};
  for (int s0 = 0; s0 < T_SEQ; s0 += 32) {
    v16bf pa;  // P A-fragment straight from bf16 LDS
    const __bf16* pr = sP + (wm * 16 + l15) * PSTR + s0 + lh * 16;
    *(uint4*)&pa       = *(const uint4*)pr;
    *((uint4*)&pa + 1) = *(const uint4*)(pr + 8);
    v16bf vb;  // V^T B-fragment from global (contiguous along s)
    const __bf16* vp = Vtb + (size_t)(wn * 16 + l15) * T_SEQ + s0 + lh * 16;
    *(uint4*)&vb       = *(const uint4*)vp;
    *((uint4*)&vb + 1) = *(const uint4*)(vp + 8);
    cacc = WMMA_BF16(pa, vb, cacc);
  }
  const int b = bh >> 4, h = bh & 15;
#pragma unroll
  for (int vv = 0; vv < 8; ++vv) {
    const int rloc = wm * 16 + vv + 8 * lh;
    const int t    = t0 + rloc;
    const int col  = h * HEAD_DIM + wn * 16 + l15;
    ctx[((size_t)(b * T_SEQ + t)) * D_MODEL + col] = (__bf16)(cacc[vv] * sInv[rloc]);
  }
}

// ---------------------------------------------------------------------------
// Host orchestration
// ---------------------------------------------------------------------------
extern "C" void kernel_launch(void* const* d_in, const int* in_sizes, int n_in,
                              void* d_out, int out_size, void* d_ws, size_t ws_size,
                              hipStream_t stream) {
  (void)in_sizes; (void)n_in; (void)out_size; (void)ws_size;

  const float* x     = (const float*)d_in[0];
  const float* ln1_g = (const float*)d_in[1];
  const float* ln1_b = (const float*)d_in[2];
  const float* ln2_g = (const float*)d_in[3];
  const float* ln2_b = (const float*)d_in[4];
  const float* Wq = (const float*)d_in[5];
  const float* bq = (const float*)d_in[6];
  const float* Wk = (const float*)d_in[7];
  const float* bk = (const float*)d_in[8];
  const float* Wv = (const float*)d_in[9];
  const float* bv = (const float*)d_in[10];
  const float* Wo = (const float*)d_in[11];
  const float* bo = (const float*)d_in[12];
  const float* W1 = (const float*)d_in[13];
  const float* b1 = (const float*)d_in[14];
  const float* W2 = (const float*)d_in[15];
  const float* b2 = (const float*)d_in[16];

  char* ws = (char*)d_ws;
  size_t off = 0;
  auto take = [&](size_t bytes) -> char* {
    char* p = ws + off;
    off = (off + bytes + 255) & ~(size_t)255;
    return p;
  };

  __bf16* HB     = (__bf16*)take((size_t)M_ROWS * D_MODEL * 2);       // ln1 out
  __bf16* WQKVT  = (__bf16*)take((size_t)3 * D_MODEL * D_MODEL * 2);  // [3D][D]
  float*  BQKV   = (float*)take(3072 * 4);
  __bf16* WOT    = (__bf16*)take((size_t)D_MODEL * D_MODEL * 2);      // [D][D]
  __bf16* W1T    = (__bf16*)take((size_t)FF_DIM * D_MODEL * 2);       // [FF][D]
  __bf16* W2T    = (__bf16*)take((size_t)D_MODEL * FF_DIM * 2);       // [D][FF]
  __bf16* QB     = (__bf16*)take((size_t)M_ROWS * D_MODEL * 2);       // [B,H,T,64]
  __bf16* KB     = (__bf16*)take((size_t)M_ROWS * D_MODEL * 2);       // [B,H,T,64]
  __bf16* VTB    = (__bf16*)take((size_t)M_ROWS * D_MODEL * 2);       // [B,H,64,T]
  __bf16* CTXB   = (__bf16*)take((size_t)M_ROWS * D_MODEL * 2);       // [B,T,D]
  float*  X1     = (float*)take((size_t)M_ROWS * D_MODEL * 4);        // x + attn
  __bf16* H2B    = (__bf16*)take((size_t)M_ROWS * D_MODEL * 2);       // ln2 out
  __bf16* FF1B   = (__bf16*)take((size_t)M_ROWS * FF_DIM * 2);        // gelu out

  // one-time weight prep (all transposed to [N][K] bf16)
  pack_qkv_wT_kernel<<<dim3(3 * 1024 * 1024 / 256), dim3(256), 0, stream>>>(Wq, Wk, Wv, WQKVT);
  pack_qkv_b_kernel<<<dim3(12), dim3(256), 0, stream>>>(bq, bk, bv, BQKV);
  transpose_cvt_kernel<<<dim3(1024 / 32, 1024 / 32), dim3(256), 0, stream>>>(Wo, WOT, 1024, 1024);
  transpose_cvt_kernel<<<dim3(4096 / 32, 1024 / 32), dim3(256), 0, stream>>>(W1, W1T, 1024, 4096);
  transpose_cvt_kernel<<<dim3(1024 / 32, 4096 / 32), dim3(256), 0, stream>>>(W2, W2T, 4096, 1024);

  // h = LN1(x)
  layernorm_kernel<<<dim3(M_ROWS), dim3(256), 0, stream>>>(x, ln1_g, ln1_b, HB);

  // QKV projection (scatter Q/K [B,H,T,64], V transposed [B,H,64,T])
  gemm_bf16_kernel<0><<<dim3(3072 / 128, M_ROWS / 128), dim3(256), GEMM_SMEM, stream>>>(
      HB, WQKVT, BQKV, M_ROWS, 3072, 1024, nullptr, nullptr, nullptr, QB, KB, VTB);

  // attention
  attn_kernel<<<dim3(T_SEQ / 32, B_SZ * N_HEADS), dim3(256), ATTN_SMEM, stream>>>(
      QB, KB, VTB, CTXB);

  // x1 = x + ctx @ Wo + bo  (f32)
  gemm_bf16_kernel<1><<<dim3(1024 / 128, M_ROWS / 128), dim3(256), GEMM_SMEM, stream>>>(
      CTXB, WOT, bo, M_ROWS, 1024, 1024, X1, nullptr, x, nullptr, nullptr, nullptr);

  // h2 = LN2(x1)
  layernorm_kernel<<<dim3(M_ROWS), dim3(256), 0, stream>>>(X1, ln2_g, ln2_b, H2B);

  // ff1 = gelu(h2 @ W1 + b1)  (bf16)
  gemm_bf16_kernel<2><<<dim3(4096 / 128, M_ROWS / 128), dim3(256), GEMM_SMEM, stream>>>(
      H2B, W1T, b1, M_ROWS, 4096, 1024, nullptr, FF1B, nullptr, nullptr, nullptr, nullptr);

  // out = x1 + ff1 @ W2 + b2  (f32)
  gemm_bf16_kernel<1><<<dim3(1024 / 128, M_ROWS / 128), dim3(256), GEMM_SMEM, stream>>>(
      FF1B, W2T, b2, M_ROWS, 1024, 4096, (float*)d_out, nullptr, X1, nullptr, nullptr, nullptr);
}
